// DTModel_89455578841552
// MI455X (gfx1250) — matrix operand order repaired and checked by reading, not compiled, not run on hardware
//
#include <hip/hip_runtime.h>
#include <hip/hip_bf16.h>
#include <math.h>

// ---------------------------------------------------------------------------
// Decision Transformer forward on gfx1250 (MI455X), bf16 WMMA everywhere.
// B=8 T=256 T3=768 H=1024 NH=16 dh=64 L=6 NI=4096
// All GEMM shapes padded so M%128==0, N%128==0, K%32==0 -> branch-free loops.
// A-operand staging uses the CDNA5 async global->LDS engine (ASYNCcnt).
// ---------------------------------------------------------------------------

typedef __bf16 bf16;
typedef __attribute__((ext_vector_type(8)))  __bf16 bf16x8;
typedef __attribute__((ext_vector_type(16))) __bf16 bf16x16;
typedef __attribute__((ext_vector_type(8)))  float  f32x8;

#define NEG_BIG (-10000.0f)

__device__ __forceinline__ bf16 f2bf(float f) {
  union { float f; unsigned int u; } v; v.f = f;
  unsigned int r = v.u + 0x7FFFu + ((v.u >> 16) & 1u);   // round-to-nearest-even
  union { unsigned short u; bf16 h; } o; o.u = (unsigned short)(r >> 16);
  return o.h;
}
__device__ __forceinline__ bf16 bf_zero() {
  union { unsigned short u; bf16 h; } o; o.u = 0; return o.h;
}

__device__ __forceinline__ bf16x16 cat16(bf16x8 lo, bf16x8 hi) {
  return __builtin_shufflevector(lo, hi, 0,1,2,3,4,5,6,7,8,9,10,11,12,13,14,15);
}

// A fragment (16x32 bf16): lane holds row (lane&15); element e -> K=(e<8?0:16)+(e&7)+8*half
__device__ __forceinline__ bf16x16 frag_a(const bf16* rowp, int half) {
  bf16x8 lo = *(const bf16x8*)(rowp + 8 * half);
  bf16x8 hi = *(const bf16x8*)(rowp + 16 + 8 * half);
  return cat16(lo, hi);
}
// B fragment (32x16 bf16): lane holds col (lane&15); element e -> K=(e<8?0:16)+(e&7)+8*half
__device__ __forceinline__ bf16x16 frag_b(const bf16* colp, int half) {
  bf16x8 lo = *(const bf16x8*)(colp + 8 * half);
  bf16x8 hi = *(const bf16x8*)(colp + 16 + 8 * half);
  return cat16(lo, hi);
}
__device__ __forceinline__ f32x8 wmma_bf(bf16x16 a, bf16x16 b, f32x8 c) {
  return __builtin_amdgcn_wmma_f32_16x16x32_bf16(false, a, false, b, (short)0, c,
                                                 false, false);
}

// --- CDNA5 async global->LDS copy (ASYNCcnt) -------------------------------
// Copies 32 bytes per lane: two GLOBAL_LOAD_ASYNC_TO_LDS_B128; INST_OFFSET is
// added to both the memory and the LDS address (ISA 15.18.3 pseudocode).
// lds_off = LDS byte offset (low 32 bits of the flat LDS pointer).
__device__ __forceinline__ void async_ld32(unsigned lds_off, const void* gptr) {
  asm volatile(
      "global_load_async_to_lds_b128 %0, %1, off\n\t"
      "global_load_async_to_lds_b128 %0, %1, off offset:16"
      :: "v"(lds_off), "v"((unsigned long long)(size_t)gptr)
      : "memory");
}
__device__ __forceinline__ void wait_async0() {
  asm volatile("s_wait_asynccnt 0" ::: "memory");
}

// ---------------------------------------------------------------------------
// bf16 GEMM:  out = epilogue(A[MxK] @ B[KxN] + bias)
// block tile 128x128, k-step 32, 8 waves: wave (wm,wn) owns 32x64.
// Requires M%128==0, N%128==0, K%32==0.
// A tile: async global->LDS, double buffered. B tile: register transpose.
// SCATTER: out_row = (m/tok_mod)*batch_stride + (m%tok_mod)*row_mul + row_off
// ---------------------------------------------------------------------------
template<bool GELU, bool RES, bool OUTF, bool OUTH, bool SCATTER>
__global__ __launch_bounds__(256) void gemm_bf16_k(
    const bf16* __restrict__ A, const bf16* __restrict__ B,
    const float* __restrict__ bias, const float* __restrict__ residual,
    float* __restrict__ outF, bf16* __restrict__ outH,
    int M, int N, int K,
    int tok_mod, int batch_stride, int row_mul, int row_off)
{
  __shared__ __align__(16) bf16 sA[2][128 * 40];  // [row][k], pad 40, double buffer
  __shared__ __align__(16) bf16 sB[128 * 40];     // [n][k]  (K-transposed), pad 40

  const int tid  = threadIdx.x;
  const int lane = tid & 31, wid = tid >> 5;
  const int half = lane >> 4, l16 = lane & 15;
  const int wm = wid & 3, wn = wid >> 2;
  const int mbase = blockIdx.y * 128, nbase = blockIdx.x * 128;

  const int arow = tid >> 1, ako = (tid & 1) * 16;   // A loader: 32B per thread
  const int bn = tid & 127, bkh = tid >> 7;          // B loader: 16 k-values at fixed n

  f32x8 acc[2][4] = {};
  alignas(16) bf16 tb[16];
  const int ktiles = K >> 5;

  const bf16* gA = A + (size_t)(mbase + arow) * K + ako;
  const bf16* gB = B + (size_t)bkh * 16 * N + nbase + bn;
  const size_t gBstep = (size_t)32 * N;

  const unsigned ldsA0 = (unsigned)(size_t)&sA[0][arow * 40 + ako];
  const unsigned ldsA1 = (unsigned)(size_t)&sA[1][arow * 40 + ako];

  // prefetch tile 0: A async into buffer 0, B into registers
  async_ld32(ldsA0, gA);
  gA += 32;
  #pragma unroll
  for (int i = 0; i < 16; i++) tb[i] = gB[(size_t)i * N];
  gB += gBstep;

  for (int kt = 0; kt < ktiles; kt++) {
    // commit B tile kt to LDS; A tile kt arrives via ASYNCcnt
    *(bf16x8*)&sB[bn * 40 + bkh * 16]     = *(bf16x8*)(tb);
    *(bf16x8*)&sB[bn * 40 + bkh * 16 + 8] = *(bf16x8*)(tb + 8);
    wait_async0();
    __syncthreads();

    // stream tile kt+1 while computing tile kt
    if (kt + 1 < ktiles) {
      async_ld32((kt & 1) ? ldsA0 : ldsA1, gA);
      gA += 32;
      #pragma unroll
      for (int i = 0; i < 16; i++) tb[i] = gB[(size_t)i * N];
      gB += gBstep;
    }

    const bf16* aBuf = (kt & 1) ? &sA[1][0] : &sA[0][0];
    bf16x16 a0 = frag_a(aBuf + (wm * 32 + l16) * 40, half);
    bf16x16 a1 = frag_a(aBuf + (wm * 32 + 16 + l16) * 40, half);
    #pragma unroll
    for (int j = 0; j < 4; j++) {
      bf16x16 bb = frag_b(&sB[(wn * 64 + j * 16 + l16) * 40], half);
      acc[0][j] = wmma_bf(a0, bb, acc[0][j]);
      acc[1][j] = wmma_bf(a1, bb, acc[1][j]);
    }
    __syncthreads();
  }

  // epilogue
  #pragma unroll
  for (int mi = 0; mi < 2; mi++) {
    #pragma unroll
    for (int j = 0; j < 4; j++) {
      const int col = nbase + wn * 64 + j * 16 + l16;
      const float bvv = bias[col];
      #pragma unroll
      for (int r = 0; r < 8; r++) {
        const int m = mbase + wm * 32 + mi * 16 + r + 8 * half;
        float vv = acc[mi][j][r] + bvv;
        if constexpr (GELU) vv = 0.5f * vv * (1.f + erff(vv * 0.70710678118654752f));
        int orow = m;
        if constexpr (SCATTER)
          orow = (m / tok_mod) * batch_stride + (m % tok_mod) * row_mul + row_off;
        if constexpr (RES) vv += residual[(size_t)orow * N + col];
        if constexpr (OUTF) outF[(size_t)orow * N + col] = vv;
        if constexpr (OUTH) outH[(size_t)orow * N + col] = f2bf(vv);
      }
    }
  }
}

// ---------------------------------------------------------------------------
// Flash attention, wave32 WMMA. Grid: (qb=6, head=16, b=8). 256 threads.
// Each wave owns 16 query rows (rows of a C-fragment stay inside one 16-lane
// half => softmax row reductions are pure intra-half shuffles).
// q/k/v point into the packed QKV buffer; ld = its row stride (3072).
// K tile staged with async global->LDS; V transposed through registers.
// ---------------------------------------------------------------------------
__global__ __launch_bounds__(256) void attn_k(
    const bf16* __restrict__ q, const bf16* __restrict__ k,
    const bf16* __restrict__ v, const float* __restrict__ amask,
    bf16* __restrict__ y, int ld)
{
  __shared__ __align__(16) bf16 sK[128 * 72];      // [key][d]   (B-frag for Q@K^T)
  __shared__ __align__(16) bf16 sVt[64 * 136];     // [d][key]   (B-frag for P@V)
  __shared__ __align__(16) bf16 sP[8 * 16 * 136];  // per-wave P 16x128 (A-frag staging)
  __shared__ float sPad[128];

  const int qb = blockIdx.x, head = blockIdx.y, b = blockIdx.z;
  const int tid = threadIdx.x;
  const int lane = tid & 31, wid = tid >> 5;
  const int half = lane >> 4, l16 = lane & 15;
  const int qrow0 = qb * 128 + wid * 16;                      // row within [0,768)
  const size_t base = ((size_t)b * 768) * ld + (size_t)head * 64;

  // Q fragments stay in registers for the whole kernel
  bf16x16 qf[2];
  #pragma unroll
  for (int ks = 0; ks < 2; ks++) {
    const bf16* gp = q + base + (size_t)(qrow0 + l16) * ld + ks * 32;
    qf[ks] = cat16(*(const bf16x8*)(gp + 8 * half),
                   *(const bf16x8*)(gp + 16 + 8 * half));
  }

  f32x8 o[4] = {};
  float mr[8], lr[8];
  #pragma unroll
  for (int r = 0; r < 8; r++) { mr[r] = -3.0e38f; lr[r] = 0.f; }

  const int ldkey = tid >> 1, lddo = (tid & 1) * 32;
  const unsigned ldsKoff = (unsigned)(size_t)&sK[ldkey * 72 + lddo];

  for (int kb = 0; kb <= qb; kb++) {
    __syncthreads();
    {
      // K rows: 64B per thread via async global->LDS
      const bf16* gk = k + base + (size_t)(kb * 128 + ldkey) * ld + lddo;
      async_ld32(ldsKoff, gk);
      async_ld32(ldsKoff + 32, gk + 16);
      // V rows: register transpose into [d][key]
      const bf16* gv = v + base + (size_t)(kb * 128 + ldkey) * ld + lddo;
      alignas(16) bf16 tv[32];
      *(bf16x8*)(tv)      = *(const bf16x8*)(gv);
      *(bf16x8*)(tv + 8)  = *(const bf16x8*)(gv + 8);
      *(bf16x8*)(tv + 16) = *(const bf16x8*)(gv + 16);
      *(bf16x8*)(tv + 24) = *(const bf16x8*)(gv + 24);
      #pragma unroll
      for (int i = 0; i < 32; i++) sVt[(lddo + i) * 136 + ldkey] = tv[i];
      if (tid < 128)
        sPad[tid] = (1.f - amask[b * 256 + (kb * 128 + tid) / 3]) * NEG_BIG;
    }
    wait_async0();
    __syncthreads();

    // S = Q K^T  (16 x 128 per wave)
    f32x8 s[8] = {};
    #pragma unroll
    for (int ks = 0; ks < 2; ks++) {
      #pragma unroll
      for (int nj = 0; nj < 8; nj++) {
        bf16x16 bb = frag_b(&sK[(nj * 16 + l16) * 72 + ks * 32], half);
        s[nj] = wmma_bf(qf[ks], bb, s[nj]);
      }
    }

    // scale + causal + pad, then online softmax
    float rowmax[8];
    #pragma unroll
    for (int r = 0; r < 8; r++) {
      const int qrow = qrow0 + r + 8 * half;
      float mx = -3.0e38f;
      #pragma unroll
      for (int nj = 0; nj < 8; nj++) {
        const int key = kb * 128 + nj * 16 + l16;
        float val = s[nj][r] * 0.125f;
        if (key > qrow) val = NEG_BIG;
        val += sPad[nj * 16 + l16];
        s[nj][r] = val;
        mx = fmaxf(mx, val);
      }
      #pragma unroll
      for (int off = 1; off < 16; off <<= 1)
        mx = fmaxf(mx, __shfl_xor(mx, off, 32));
      rowmax[r] = mx;
    }
    float alpha[8], rsum[8];
    #pragma unroll
    for (int r = 0; r < 8; r++) {
      const float mnew = fmaxf(mr[r], rowmax[r]);
      alpha[r] = __expf(mr[r] - mnew);
      mr[r] = mnew;
      rsum[r] = 0.f;
    }
    #pragma unroll
    for (int nj = 0; nj < 8; nj++) {
      #pragma unroll
      for (int r = 0; r < 8; r++) {
        const float p = __expf(s[nj][r] - mr[r]);
        rsum[r] += p;
        sP[wid * 2176 + (r + 8 * half) * 136 + nj * 16 + l16] = f2bf(p);
      }
    }
    #pragma unroll
    for (int r = 0; r < 8; r++) {
      float t = rsum[r];
      #pragma unroll
      for (int off = 1; off < 16; off <<= 1) t += __shfl_xor(t, off, 32);
      lr[r] = lr[r] * alpha[r] + t;
    }
    #pragma unroll
    for (int c = 0; c < 4; c++)
      #pragma unroll
      for (int r = 0; r < 8; r++) o[c][r] *= alpha[r];

    // O += P V  (contraction over 128 keys = 4 k-steps)
    #pragma unroll
    for (int kk = 0; kk < 4; kk++) {
      bf16x16 af = frag_a(&sP[wid * 2176 + l16 * 136 + kk * 32], half);
      #pragma unroll
      for (int c = 0; c < 4; c++) {
        bf16x16 bb = frag_b(&sVt[(c * 16 + l16) * 136 + kk * 32], half);
        o[c] = wmma_bf(af, bb, o[c]);
      }
    }
  }

  float inv[8];
  #pragma unroll
  for (int r = 0; r < 8; r++) inv[r] = 1.f / lr[r];
  const size_t ybase = ((size_t)b * 768) * 1024 + (size_t)head * 64;
  #pragma unroll
  for (int c = 0; c < 4; c++) {
    #pragma unroll
    for (int r = 0; r < 8; r++) {
      const int row = qrow0 + r + 8 * half;
      y[ybase + (size_t)row * 1024 + c * 16 + l16] = f2bf(o[c][r] * inv[r]);
    }
  }
}

// ---------------------------------------------------------------------------
// LayerNorm over rows of 1024; writes f32 and/or bf16.
// ---------------------------------------------------------------------------
__global__ __launch_bounds__(256) void ln_k(
    const float* __restrict__ x, const float* __restrict__ g,
    const float* __restrict__ bta, float* __restrict__ outF,
    bf16* __restrict__ outH)
{
  const int row = blockIdx.x, tid = threadIdx.x;
  const float* xr = x + (size_t)row * 1024;
  float vbuf[4];
  float s = 0.f;
  #pragma unroll
  for (int i = 0; i < 4; i++) { vbuf[i] = xr[tid + 256 * i]; s += vbuf[i]; }
  __shared__ float red[256];
  red[tid] = s; __syncthreads();
  for (int st = 128; st > 0; st >>= 1) { if (tid < st) red[tid] += red[tid + st]; __syncthreads(); }
  const float mean = red[0] * (1.f / 1024.f);
  __syncthreads();
  float s2 = 0.f;
  #pragma unroll
  for (int i = 0; i < 4; i++) { float d = vbuf[i] - mean; s2 += d * d; }
  red[tid] = s2; __syncthreads();
  for (int st = 128; st > 0; st >>= 1) { if (tid < st) red[tid] += red[tid + st]; __syncthreads(); }
  const float rinv = rsqrtf(red[0] * (1.f / 1024.f) + 1e-5f);
  #pragma unroll
  for (int i = 0; i < 4; i++) {
    const int c = tid + 256 * i;
    const float ov = (vbuf[i] - mean) * rinv * g[c] + bta[c];
    if (outF) outF[(size_t)row * 1024 + c] = ov;
    if (outH) outH[(size_t)row * 1024 + c] = f2bf(ov);
  }
}

// ---------------------------------------------------------------------------
// Embedding stage: build the three 2048x1056 (K padded) bf16 concat matrices.
// cols 0..1023 = proj, 1024..1031 = time embedding, 1032..1055 = zero pad.
// ---------------------------------------------------------------------------
__global__ __launch_bounds__(256) void embed_build_k(
    const float* __restrict__ states, const float* __restrict__ actions,
    const float* __restrict__ rtgs, const float* __restrict__ smean,
    const float* __restrict__ sstd, const float* __restrict__ Wes,
    const float* __restrict__ bes, const float* __restrict__ Wea,
    const float* __restrict__ bea, const float* __restrict__ Wer,
    const float* __restrict__ ber, const float* __restrict__ Et,
    const int* __restrict__ tsteps,
    bf16* __restrict__ Xs, bf16* __restrict__ Xa, bf16* __restrict__ Xr)
{
  const int token = blockIdx.x, tid = threadIdx.x;
  __shared__ float sn[16];
  __shared__ float te[8];
  if (tid < 16) sn[tid] = (states[token * 16 + tid] - smean[tid]) / (sstd[tid] + 1e-9f);
  if (tid < 8) { const int ts = tsteps[token]; te[tid] = Et[ts * 8 + tid]; }
  __syncthreads();
  const float av = actions[token];
  const float rv = rtgs[token];
  const size_t rb = (size_t)token * 1056;
  for (int j = tid; j < 1024; j += 256) {
    float ss = bes[j];
    #pragma unroll
    for (int i = 0; i < 16; i++) ss += sn[i] * Wes[i * 1024 + j];
    Xs[rb + j] = f2bf(ss);
    Xa[rb + j] = f2bf(bea[j] + av * Wea[j]);
    Xr[rb + j] = f2bf(ber[j] + rv * Wer[j]);
  }
  if (tid < 32) {
    const bf16 tvb = (tid < 8) ? f2bf(te[tid]) : bf_zero();
    Xs[rb + 1024 + tid] = tvb;
    Xa[rb + 1024 + tid] = tvb;
    Xr[rb + 1024 + tid] = tvb;
  }
}

// ---------------------------------------------------------------------------
// Action head: out[b,t] = x[b, 3t+1, :] . W_pa + b_pa
// ---------------------------------------------------------------------------
__global__ __launch_bounds__(256) void head_k(
    const float* __restrict__ x, const float* __restrict__ Wpa,
    const float* __restrict__ bpa, float* __restrict__ out)
{
  const int token = blockIdx.x, tid = threadIdx.x;
  const int b = token >> 8, t = token & 255;
  const float* xr = x + ((size_t)b * 768 + t * 3 + 1) * 1024;
  float s = 0.f;
  for (int i = tid; i < 1024; i += 256) s += xr[i] * Wpa[i];
  __shared__ float red[256];
  red[tid] = s; __syncthreads();
  for (int st = 128; st > 0; st >>= 1) { if (tid < st) red[tid] += red[tid + st]; __syncthreads(); }
  if (tid == 0) out[token] = red[0] + bpa[0];
}

// f32 -> bf16 contiguous
__global__ __launch_bounds__(256) void cvt_k(const float* __restrict__ src,
                                             bf16* __restrict__ dst, long n)
{
  const long i = (long)blockIdx.x * 256 + threadIdx.x;
  if (i < n) dst[i] = f2bf(src[i]);
}

// f32 (srcRows x cols) -> bf16 (dstRows x cols), extra rows zero-filled
__global__ __launch_bounds__(256) void cvt_pad_k(const float* __restrict__ src,
                                                 bf16* __restrict__ dst,
                                                 int srcRows, int cols, long n)
{
  const long i = (long)blockIdx.x * 256 + threadIdx.x;
  if (i < n) {
    const long r = i / cols;
    dst[i] = (r < srcRows) ? f2bf(src[i]) : bf_zero();
  }
}

// f32 (rows x cols) -> bf16 at dst[row*dstride + coff + col]  (QKV packing)
__global__ __launch_bounds__(256) void cvt_off_k(const float* __restrict__ src,
                                                 bf16* __restrict__ dst,
                                                 int cols, int dstride, int coff,
                                                 long n)
{
  const long i = (long)blockIdx.x * 256 + threadIdx.x;
  if (i < n) {
    const long r = i / cols, c = i - r * cols;
    dst[r * dstride + coff + c] = f2bf(src[i]);
  }
}

// pack [bq|bk|bv] (each 1024) into one 3072 f32 vector
__global__ __launch_bounds__(256) void pack3_k(const float* __restrict__ a,
                                               const float* __restrict__ b,
                                               const float* __restrict__ c,
                                               float* __restrict__ dst)
{
  const int i = blockIdx.x * 256 + threadIdx.x;
  if (i < 1024) dst[i] = a[i];
  else if (i < 2048) dst[i] = b[i - 1024];
  else if (i < 3072) dst[i] = c[i - 2048];
}

// ---------------------------------------------------------------------------
extern "C" void kernel_launch(void* const* d_in, const int* in_sizes, int n_in,
                              void* d_out, int out_size, void* d_ws, size_t ws_size,
                              hipStream_t stream)
{
  const float* states = (const float*)d_in[0];
  const float* actions = (const float*)d_in[1];
  const float* rtgs   = (const float*)d_in[2];
  const float* amask  = (const float*)d_in[3];
  const float* smean  = (const float*)d_in[4];
  const float* sstd   = (const float*)d_in[5];
  const float* Wes = (const float*)d_in[6];
  const float* bes = (const float*)d_in[7];
  const float* Wea = (const float*)d_in[8];
  const float* bea = (const float*)d_in[9];
  const float* Wer = (const float*)d_in[10];
  const float* ber = (const float*)d_in[11];
  const float* Et  = (const float*)d_in[12];
  const float* Wts = (const float*)d_in[13];
  const float* bts = (const float*)d_in[14];
  const float* Wta = (const float*)d_in[15];
  const float* btaP= (const float*)d_in[16];
  const float* Wtr = (const float*)d_in[17];
  const float* btr = (const float*)d_in[18];
  const float* eg  = (const float*)d_in[19];
  const float* eb  = (const float*)d_in[20];
  const float* l1g = (const float*)d_in[21];
  const float* l1b = (const float*)d_in[22];
  const float* l2g = (const float*)d_in[23];
  const float* l2b = (const float*)d_in[24];
  const float* Wq  = (const float*)d_in[25];
  const float* bq  = (const float*)d_in[26];
  const float* Wk  = (const float*)d_in[27];
  const float* bk  = (const float*)d_in[28];
  const float* Wv  = (const float*)d_in[29];
  const float* bv  = (const float*)d_in[30];
  const float* Wo  = (const float*)d_in[31];
  const float* bo  = (const float*)d_in[32];
  const float* W1  = (const float*)d_in[33];
  const float* b1  = (const float*)d_in[34];
  const float* W2  = (const float*)d_in[35];
  const float* b2  = (const float*)d_in[36];
  const float* Wpa = (const float*)d_in[37];
  const float* bpa = (const float*)d_in[38];
  const int*   tst = (const int*)d_in[39];
  (void)in_sizes; (void)n_in; (void)out_size; (void)ws_size;

  char* ws = (char*)d_ws;
  size_t off = 0;
  auto wsalloc = [&](size_t bytes) -> char* {
    char* p = ws + off; off += (bytes + 255) & ~(size_t)255; return p;
  };

  float* x   = (float*)wsalloc((size_t)6144 * 1024 * 4);
  bf16* h    = (bf16*)wsalloc((size_t)6144 * 1024 * 2);
  bf16* qkv  = (bf16*)wsalloc((size_t)6144 * 3072 * 2);
  bf16* yb   = (bf16*)wsalloc((size_t)6144 * 1024 * 2);
  bf16* mid  = (bf16*)wsalloc((size_t)6144 * 4096 * 2);
  bf16* wqkvS= (bf16*)wsalloc((size_t)1024 * 3072 * 2);
  bf16* woS  = (bf16*)wsalloc((size_t)1024 * 1024 * 2);
  bf16* w1S  = (bf16*)wsalloc((size_t)1024 * 4096 * 2);
  bf16* w2S  = (bf16*)wsalloc((size_t)4096 * 1024 * 2);
  float* bqkv= (float*)wsalloc((size_t)3072 * 4);
  bf16* wtsS = (bf16*)wsalloc((size_t)1056 * 1024 * 2);
  bf16* wtaS = (bf16*)wsalloc((size_t)1056 * 1024 * 2);
  bf16* wtrS = (bf16*)wsalloc((size_t)1056 * 1024 * 2);
  bf16* Xs   = (bf16*)wsalloc((size_t)2048 * 1056 * 2);
  bf16* Xa   = (bf16*)wsalloc((size_t)2048 * 1056 * 2);
  bf16* Xr   = (bf16*)wsalloc((size_t)2048 * 1056 * 2);

  auto cvt = [&](const float* s, bf16* d, long n) {
    cvt_k<<<dim3((unsigned)((n + 255) / 256)), dim3(256), 0, stream>>>(s, d, n);
  };

  // --- embedding stage -----------------------------------------------------
  {
    const long n = 1056L * 1024;
    cvt_pad_k<<<dim3((unsigned)((n + 255) / 256)), 256, 0, stream>>>(Wts, wtsS, 1032, 1024, n);
    cvt_pad_k<<<dim3((unsigned)((n + 255) / 256)), 256, 0, stream>>>(Wta, wtaS, 1032, 1024, n);
    cvt_pad_k<<<dim3((unsigned)((n + 255) / 256)), 256, 0, stream>>>(Wtr, wtrS, 1032, 1024, n);
  }
  embed_build_k<<<dim3(2048), dim3(256), 0, stream>>>(
      states, actions, rtgs, smean, sstd, Wes, bes, Wea, bea, Wer, ber, Et, tst,
      Xs, Xa, Xr);

  dim3 gEmb(1024 / 128, 2048 / 128);
  // scatter rows into x as [re, se, ae] interleave: orow=(m/256)*768+(m%256)*3+off
  gemm_bf16_k<false,false,true,false,true><<<gEmb, 256, 0, stream>>>(
      Xr, wtrS, btr, nullptr, x, nullptr, 2048, 1024, 1056, 256, 768, 3, 0);
  gemm_bf16_k<false,false,true,false,true><<<gEmb, 256, 0, stream>>>(
      Xs, wtsS, bts, nullptr, x, nullptr, 2048, 1024, 1056, 256, 768, 3, 1);
  gemm_bf16_k<false,false,true,false,true><<<gEmb, 256, 0, stream>>>(
      Xa, wtaS, btaP, nullptr, x, nullptr, 2048, 1024, 1056, 256, 768, 3, 2);
  ln_k<<<dim3(6144), 256, 0, stream>>>(x, eg, eb, x, nullptr);   // embedding LN

  // --- transformer layers --------------------------------------------------
  dim3 gQKV(3072 / 128, 6144 / 128);
  dim3 gMain(1024 / 128, 6144 / 128);
  dim3 gMlp1(4096 / 128, 6144 / 128);
  for (int l = 0; l < 6; l++) {
    {
      const long n = 1024L * 1024;
      const unsigned g = (unsigned)((n + 255) / 256);
      cvt_off_k<<<g, 256, 0, stream>>>(Wq + (size_t)l * n, wqkvS, 1024, 3072, 0,    n);
      cvt_off_k<<<g, 256, 0, stream>>>(Wk + (size_t)l * n, wqkvS, 1024, 3072, 1024, n);
      cvt_off_k<<<g, 256, 0, stream>>>(Wv + (size_t)l * n, wqkvS, 1024, 3072, 2048, n);
    }
    cvt(Wo + (size_t)l * 1024 * 1024, woS, 1024L * 1024);
    cvt(W1 + (size_t)l * 1024 * 4096, w1S, 1024L * 4096);
    cvt(W2 + (size_t)l * 4096 * 1024, w2S, 4096L * 1024);
    pack3_k<<<dim3(12), 256, 0, stream>>>(bq + l * 1024, bk + l * 1024, bv + l * 1024, bqkv);

    ln_k<<<dim3(6144), 256, 0, stream>>>(x, l1g + l * 1024, l1b + l * 1024,
                                         nullptr, h);

    // fused QKV projection: [6144 x 1024] @ [1024 x 3072]
    gemm_bf16_k<false,false,false,true,false><<<gQKV, 256, 0, stream>>>(
        h, wqkvS, bqkv, nullptr, nullptr, qkv, 6144, 3072, 1024, 0, 0, 1, 0);

    attn_k<<<dim3(6, 16, 8), 256, 0, stream>>>(qkv, qkv + 1024, qkv + 2048,
                                               amask, yb, 3072);

    gemm_bf16_k<false,true,true,false,false><<<gMain, 256, 0, stream>>>(
        yb, woS, bo + l * 1024, x, x, nullptr, 6144, 1024, 1024, 0, 0, 1, 0);

    ln_k<<<dim3(6144), 256, 0, stream>>>(x, l2g + l * 1024, l2b + l * 1024,
                                         nullptr, h);

    gemm_bf16_k<true,false,false,true,false><<<gMlp1, 256, 0, stream>>>(
        h, w1S, b1 + (size_t)l * 4096, nullptr, nullptr, mid, 6144, 4096, 1024, 0, 0, 1, 0);
    gemm_bf16_k<false,true,true,false,false><<<gMain, 256, 0, stream>>>(
        mid, w2S, b2 + l * 1024, x, x, nullptr, 6144, 1024, 4096, 0, 0, 1, 0);
  }

  head_k<<<dim3(2048), 256, 0, stream>>>(x, Wpa, bpa, (float*)d_out);
}
